// Head_43963285241944
// MI455X (gfx1250) — matrix operand order repaired and checked by reading, not compiled, not run on hardware
//
#include <hip/hip_runtime.h>
#include <hip/hip_bf16.h>

#define D_MODEL 1024
#define HEADSZ  1024
#define SEQ     128
#define BATCH   256
#define SMS     136   // attn LDS row stride in floats
#define KSTEP   64    // proj k-elements staged per TDM DMA
#define BROW    80    // proj LDS row stride in halfs: [32 data][8 pad][32 data][8 pad]

// sched_group_barrier masks
#define SG_WMMA   0x008
#define SG_VMEMRD 0x020
#define SG_DSRD   0x100

typedef __attribute__((ext_vector_type(16))) __bf16 v16bf;
typedef __attribute__((ext_vector_type(8)))  __bf16 v8bf;
typedef __attribute__((ext_vector_type(8)))  float  v8f;
typedef __attribute__((ext_vector_type(4)))  float  v4f;
typedef __attribute__((ext_vector_type(4)))  unsigned int v4u;
typedef __attribute__((ext_vector_type(8)))  int    v8i;
typedef __attribute__((ext_vector_type(4)))  int    v4i;

static __device__ __forceinline__ v8f wmma_bf16(v16bf a, v16bf b, v8f c) {
  // (neg_a, A, neg_b, B, c_mod, C, reuse_a, reuse_b)
  return __builtin_amdgcn_wmma_f32_16x16x32_bf16(false, a, false, b, (short)0, c,
                                                 false, false);
}

// TDM load of a 128 x KSTEP bf16 tile (rows n0.., cols k0.. of Wt[1024,1024]
// row-major) into LDS at lds_byte_addr.  LDS padding: 4 DWORDs every 16 DWORDs
// => row layout [32 data][8 pad][32 data][8 pad] = BROW(80) halfs per row.
static __device__ __forceinline__ void tdm_load_wt_tile(const __bf16* Wt, int n0,
                                                        int k0,
                                                        unsigned int lds_byte_addr) {
  unsigned long long ga =
      (unsigned long long)(Wt + (unsigned long long)n0 * D_MODEL + k0);
  // D# group 0: [1:0]=count=1, [63:32]=lds_addr, [120:64]=global_addr, [127:126]=type=2
  v4u g0 = {1u, lds_byte_addr, (unsigned int)ga,
            (unsigned int)(ga >> 32) | (2u << 30)};
  // D# group 1:
  //  [17:16]  data_size=1 (2 bytes)
  //  [20]     pad_enable=1, [24:22] pad_interval=3 (16 DWORDs), [31:25] pad_amount=3 (4 DWORDs)
  //  [79:48]  tensor_dim0 = 1024, [111:80] tensor_dim1 = 1024
  //  [127:112] tile_dim0 = KSTEP, [143:128] tile_dim1 = 128
  //  [207:160] tensor_dim0_stride = 1024
  v8i g1 = {(int)((1u << 16) | (1u << 20) | (3u << 22) | (3u << 25)),
            (int)(1024u << 16),            // tensor_dim0[15:0] at bits[63:48]
            (int)(1024u << 16),            // dim0 hi | tensor_dim1[15:0] at [95:80]
            (int)((unsigned)KSTEP << 16),  // dim1 hi | tile_dim0 at [127:112]
            128,                           // tile_dim1 at [143:128]
            1024,                          // tensor_dim0_stride[31:0]
            0, 0};
  v4i g2 = {0, 0, 0, 0};
  v4i g3 = {0, 0, 0, 0};
#if defined(__clang_major__) && (__clang_major__ >= 23)
  v8i g4 = {0, 0, 0, 0, 0, 0, 0, 0};
  __builtin_amdgcn_tensor_load_to_lds(g0, g1, g2, g3, g4, 0);
#else
  __builtin_amdgcn_tensor_load_to_lds(g0, g1, g2, g3, 0);
#endif
}

// Load one stage (2 chunks of 32 K-elements) of the A operand as fp32 and
// convert straight to the bf16 WMMA A layout (16 VGPRs live, raw transients).
static __device__ __forceinline__ void load_a_stage(const float* ap, v16bf out[2]) {
#pragma unroll
  for (int c = 0; c < 2; ++c) {
    const float* p = ap + c * 32;
    v4f f0 = *(const v4f*)(p);
    v4f f1 = *(const v4f*)(p + 4);
    v4f f2 = *(const v4f*)(p + 16);
    v4f f3 = *(const v4f*)(p + 20);
#pragma unroll
    for (int i = 0; i < 4; ++i) {
      out[c][i]      = (__bf16)f0[i];
      out[c][4 + i]  = (__bf16)f1[i];
      out[c][8 + i]  = (__bf16)f2[i];
      out[c][12 + i] = (__bf16)f3[i];
    }
  }
}

// One 32-wide k-chunk for this wave's 4 N-tiles: 4 B tiles from LDS + 4 WMMAs.
static __device__ __forceinline__ void proj_chunk(const __bf16* lb, int c,
                                                  int tt, int lan16, int half,
                                                  v16bf a, v8f acc[4]) {
  v8bf b0[4], b1[4];
#pragma unroll
  for (int t = 0; t < 4; ++t) {
    const __bf16* bp =
        lb + ((tt + t) * 16 + lan16) * BROW + c * 40 + half * 8;
    b0[t] = *(const v8bf*)(bp);
    b1[t] = *(const v8bf*)(bp + 16);
  }
#pragma unroll
  for (int t = 0; t < 4; ++t) {
    v16bf b;
#pragma unroll
    for (int i = 0; i < 8; ++i) { b[i] = b0[t][i]; b[8 + i] = b1[t][i]; }
    acc[t] = wmma_bf16(a, b, acc[t]);
  }
}

// One pipeline stage.  Scheduling directives prescribe:
//   [8 global A loads] -> [8 DS reads chunk0] -> 2 WMMA
//   -> [8 DS reads chunk1 overlapping remaining chunk0 WMMAs] -> 6 WMMA
static __device__ __forceinline__ void proj_stage(
    const float* ap_base, const __bf16* Wt, int n0, int ks_next, bool more,
    unsigned int lds_next, int wave, const __bf16* lb, int tt, int lan16,
    int half, v16bf acur[2], v16bf anext[2], v8f acc[4]) {
  if (wave == 0 && more) tdm_load_wt_tile(Wt, n0, ks_next, lds_next);
  proj_chunk(lb, 0, tt, lan16, half, acur[0], acc);
  if (more) load_a_stage(ap_base + ks_next, anext);
  proj_chunk(lb, 1, tt, lan16, half, acur[1], acc);
  if (more) __builtin_amdgcn_sched_group_barrier(SG_VMEMRD, 8, 0);
  __builtin_amdgcn_sched_group_barrier(SG_DSRD, 8, 0);
  __builtin_amdgcn_sched_group_barrier(SG_WMMA, 2, 0);
  __builtin_amdgcn_sched_group_barrier(SG_DSRD, 8, 0);
  __builtin_amdgcn_sched_group_barrier(SG_WMMA, 6, 0);
}

// ---------------------------------------------------------------------------
// Kernel 0: convert + transpose weights  W[f32, D x H] -> Wt[bf16, H x D]
// ---------------------------------------------------------------------------
__global__ void wt_kernel(const float* __restrict__ w, __bf16* __restrict__ wt) {
  int idx = blockIdx.x * blockDim.x + threadIdx.x;  // over H*D outputs
  int n = idx >> 10;    // row of Wt (H index)
  int d = idx & 1023;   // col of Wt (D index)
  wt[idx] = (__bf16)w[d * HEADSZ + n];
}

// ---------------------------------------------------------------------------
// Kernel 1: projection  Out = X[f32, (B*S) x D] @ Wt^T  (Wt is bf16 [H x D])
// 512 threads = 16 waves per 128x128 tile; wave w owns rows [16*(w&7),+16) x
// cols [64*(w>>3),+64) -> 4 acc tiles/wave, low register pressure.
// TDM-staged double-buffered B tile in LDS; A pipelined one stage ahead.
// Last stage pair peeled so the hot loop has compile-time 'more'.
//   v_transpose==0 : Out = bf16 [B*S, H]   (for Q and K)
//   v_transpose==1 : Out = bf16 [B, H, S]  (V stored transposed per batch)
// ---------------------------------------------------------------------------
__global__ void
__launch_bounds__(512, 1)
__attribute__((amdgpu_waves_per_eu(1)))
proj_kernel(const float* __restrict__ X, const __bf16* __restrict__ Wt,
            __bf16* __restrict__ Out, int v_transpose) {
  __shared__ __bf16 lds_b[2][128 * BROW];   // 2 x 20480 B

  const int lane  = threadIdx.x & 31;
  const int wave  = threadIdx.x >> 5;      // 0..15
  const int half  = lane >> 4;
  const int lan16 = lane & 15;
  const int mrow  = (wave & 7) * 16;       // row block within tile
  const int tt    = (wave >> 3) * 4;       // first of this wave's 4 N-tiles
  const int bidx  = blockIdx.y;            // 128-row M tile == one batch
  const int n0    = blockIdx.x * 128;
  const int m0    = bidx * 128 + mrow;

  const unsigned int lds0 = (unsigned int)(size_t)(&lds_b[0][0]);
  const unsigned int lds1 = (unsigned int)(size_t)(&lds_b[1][0]);

  v8f acc[4] = {};

  // A-layout (16-bit, 16x32): lane<16 holds K {0..7,16..23}, lane>=16 {8..15,24..31}
  const float* ap_base = X + (size_t)(m0 + lan16) * D_MODEL + half * 8;

  // Prologue: stage-0 A (converted), DMA first B tile into buffer 0.
  v16bf aA[2], aB[2];
  load_a_stage(ap_base, aA);
  if (wave == 0) {
    tdm_load_wt_tile(Wt, n0, 0, lds0);
    __builtin_amdgcn_s_wait_tensorcnt(0);
  }
  __syncthreads();

  // Main loop: 7 double-stages, prefetch always on (no branches in region).
  for (int ks = 0; ks < D_MODEL - 2 * KSTEP; ks += 2 * KSTEP) {
    proj_stage(ap_base, Wt, n0, ks + KSTEP, true, lds1, wave, &lds_b[0][0], tt,
               lan16, half, aA, aB, acc);
    if (wave == 0) __builtin_amdgcn_s_wait_tensorcnt(0);
    __syncthreads();
    proj_stage(ap_base, Wt, n0, ks + 2 * KSTEP, true, lds0, wave,
               &lds_b[1][0], tt, lan16, half, aB, aA, acc);
    if (wave == 0) __builtin_amdgcn_s_wait_tensorcnt(0);
    __syncthreads();
  }
  // Epilogue: stages 14 and 15 (ks = 896, 960).
  proj_stage(ap_base, Wt, n0, D_MODEL - KSTEP, true, lds1, wave, &lds_b[0][0],
             tt, lan16, half, aA, aB, acc);
  if (wave == 0) __builtin_amdgcn_s_wait_tensorcnt(0);
  __syncthreads();
  proj_stage(ap_base, Wt, n0, D_MODEL, false, lds0, wave, &lds_b[1][0], tt,
             lan16, half, aB, aA, acc);

  // C layout: VGPR r, lanes 0-15 -> M=r, lanes 16-31 -> M=r+8; N = lane%16
  if (!v_transpose) {
#pragma unroll
    for (int t = 0; t < 4; ++t)
#pragma unroll
      for (int r = 0; r < 8; ++r) {
        int m = m0 + r + 8 * half;
        int n = n0 + (tt + t) * 16 + lan16;
        Out[(size_t)m * HEADSZ + n] = (__bf16)acc[t][r];
      }
  } else {
#pragma unroll
    for (int t = 0; t < 4; ++t)
#pragma unroll
      for (int r = 0; r < 8; ++r) {
        int s = mrow + r + 8 * half;            // sequence position within batch
        int n = n0 + (tt + t) * 16 + lan16;     // head dim
        Out[((size_t)bidx * HEADSZ + n) * SEQ + s] = (__bf16)acc[t][r];
      }
  }
}

// ---------------------------------------------------------------------------
// Kernel 2: attention per batch.  scores = Q Kt / 32, causal softmax, P @ V.
// Block = one batch, 8 waves; wave w owns query rows [16w,16w+16).
// ---------------------------------------------------------------------------
__global__ void attn_kernel(const __bf16* __restrict__ Q,
                            const __bf16* __restrict__ K,
                            const __bf16* __restrict__ Vt,
                            float* __restrict__ Out,
                            const int* __restrict__ maskp) {
  extern __shared__ float smem[];   // SEQ x SMS floats
  const int lane  = threadIdx.x & 31;
  const int wave  = threadIdx.x >> 5;
  const int half  = lane >> 4;
  const int lan16 = lane & 15;
  const int b     = blockIdx.x;
  const int m0    = wave * 16;
  const int mask_on = *maskp;

  const __bf16* Qb = Q  + (size_t)b * SEQ * HEADSZ;
  const __bf16* Kb = K  + (size_t)b * SEQ * HEADSZ;
  const __bf16* Vb = Vt + (size_t)b * HEADSZ * SEQ;

  // ---- scores = Q @ K^T : both operands contiguous row loads (NT gemm) ----
  v8f sc[8] = {};
  const __bf16* qp_base = Qb + (size_t)(m0 + lan16) * HEADSZ + half * 16;
  for (int k0 = 0; k0 < HEADSZ; k0 += 32) {
    // Preload all 8 K tiles (32B-aligned direct v16bf loads).
    v16bf bt[8];
#pragma unroll
    for (int t = 0; t < 8; ++t)
      bt[t] = *(const v16bf*)(Kb + (size_t)(t * 16 + lan16) * HEADSZ + k0 +
                              half * 16);
    v8bf c0 = *(const v8bf*)(qp_base + k0);
    v8bf c1 = *(const v8bf*)(qp_base + k0 + 8);
    v16bf a;
#pragma unroll
    for (int i = 0; i < 8; ++i) { a[i] = c0[i]; a[8 + i] = c1[i]; }
#pragma unroll
    for (int t = 0; t < 8; ++t) sc[t] = wmma_bf16(a, bt[t], sc[t]);
    __builtin_amdgcn_sched_group_barrier(SG_VMEMRD, 18, 0);
    __builtin_amdgcn_sched_group_barrier(SG_WMMA, 8, 0);
  }

  // ---- scale + causal mask, stage to LDS ----
  const float scale = 0.03125f;  // 1/sqrt(1024)
#pragma unroll
  for (int t = 0; t < 8; ++t)
#pragma unroll
    for (int r = 0; r < 8; ++r) {
      int m = m0 + r + 8 * half;
      int n = t * 16 + lan16;
      float s = sc[t][r] * scale;
      if (mask_on && n > m) s = -__builtin_inff();
      smem[m * SMS + n] = s;
    }
  __syncthreads();

  // ---- row softmax: one thread per row ----
  if (threadIdx.x < SEQ) {
    float* row = smem + threadIdx.x * SMS;
    float mx = row[0];
    for (int j = 1; j < SEQ; ++j) mx = fmaxf(mx, row[j]);
    float sum = 0.0f;
    for (int j = 0; j < SEQ; ++j) { float e = __expf(row[j] - mx); row[j] = e; sum += e; }
    float inv = 1.0f / sum;
    for (int j = 0; j < SEQ; ++j) row[j] *= inv;
  }
  __syncthreads();

  // ---- hoist attn A-tiles (K = 128 -> 4 steps of 32) from LDS, cvt to bf16 ----
  v16bf aa[4];
#pragma unroll
  for (int kk = 0; kk < 4; ++kk) {
    const float* rp = smem + (m0 + lan16) * SMS + kk * 32 + half * 16;
    v4f f0 = *(const v4f*)(rp);
    v4f f1 = *(const v4f*)(rp + 4);
    v4f f2 = *(const v4f*)(rp + 8);
    v4f f3 = *(const v4f*)(rp + 12);
#pragma unroll
    for (int i = 0; i < 4; ++i) {
      aa[kk][i]      = (__bf16)f0[i];
      aa[kk][4 + i]  = (__bf16)f1[i];
      aa[kk][8 + i]  = (__bf16)f2[i];
      aa[kk][12 + i] = (__bf16)f3[i];
    }
  }

  // ---- Out = P @ V : B operand from transposed V (contiguous row loads) ----
  float* Ob = Out + (size_t)b * SEQ * HEADSZ;
  for (int nc = 0; nc < HEADSZ; nc += 128) {
    v8f acc[8] = {};
#pragma unroll
    for (int kk = 0; kk < 4; ++kk) {
#pragma unroll
      for (int g = 0; g < 2; ++g) {          // groups of 4 to bound liveness
        v16bf bt[4];
#pragma unroll
        for (int t = 0; t < 4; ++t)
          bt[t] = *(const v16bf*)(Vb +
                                  (size_t)(nc + (g * 4 + t) * 16 + lan16) * SEQ +
                                  kk * 32 + half * 16);
#pragma unroll
        for (int t = 0; t < 4; ++t)
          acc[g * 4 + t] = wmma_bf16(aa[kk], bt[t], acc[g * 4 + t]);
      }
    }
#pragma unroll
    for (int t = 0; t < 8; ++t)
#pragma unroll
      for (int r = 0; r < 8; ++r) {
        int m = m0 + r + 8 * half;
        int n = nc + t * 16 + lan16;
        Ob[(size_t)m * HEADSZ + n] = acc[t][r];
      }
  }
}

// ---------------------------------------------------------------------------
extern "C" void kernel_launch(void* const* d_in, const int* in_sizes, int n_in,
                              void* d_out, int out_size, void* d_ws, size_t ws_size,
                              hipStream_t stream) {
  const float* query = (const float*)d_in[0];
  const float* key_  = (const float*)d_in[1];
  const float* value = (const float*)d_in[2];
  const float* w_q   = (const float*)d_in[3];
  const float* w_k   = (const float*)d_in[4];
  const float* w_v   = (const float*)d_in[5];
  const int*   maskp = (const int*)d_in[6];

  // workspace layout (bf16): 3 transposed weights + q + k + v_transposed
  __bf16* wtq = (__bf16*)d_ws;
  __bf16* wtk = wtq + (size_t)D_MODEL * HEADSZ;
  __bf16* wtv = wtk + (size_t)D_MODEL * HEADSZ;
  __bf16* qb  = wtv + (size_t)D_MODEL * HEADSZ;
  __bf16* kb  = qb  + (size_t)BATCH * SEQ * HEADSZ;
  __bf16* vtb = kb  + (size_t)BATCH * SEQ * HEADSZ;

  dim3 tb(256);
  wt_kernel<<<dim3((D_MODEL * HEADSZ) / 256), tb, 0, stream>>>(w_q, wtq);
  wt_kernel<<<dim3((D_MODEL * HEADSZ) / 256), tb, 0, stream>>>(w_k, wtk);
  wt_kernel<<<dim3((D_MODEL * HEADSZ) / 256), tb, 0, stream>>>(w_v, wtv);

  proj_kernel<<<dim3(HEADSZ / 128, BATCH), dim3(512), 0, stream>>>(query, wtq, qb, 0);
  proj_kernel<<<dim3(HEADSZ / 128, BATCH), dim3(512), 0, stream>>>(key_,  wtk, kb, 0);
  proj_kernel<<<dim3(HEADSZ / 128, BATCH), dim3(512), 0, stream>>>(value, wtv, vtb, 1);

  attn_kernel<<<dim3(BATCH), tb, (size_t)(SEQ * SMS * sizeof(float)), stream>>>(
      qb, kb, vtb, (float*)d_out, maskp);
}
